// relation_module_2740189135439
// MI455X (gfx1250) — compile-verified, matrix operand order checked by reading, not run on hardware
//
#include <hip/hip_runtime.h>

typedef __attribute__((ext_vector_type(16))) __bf16 v16bf;
typedef __attribute__((ext_vector_type(8)))  __bf16 v8bf;
typedef __attribute__((ext_vector_type(8)))  float  v8f;

#define EM_DIM 64
#define GROUP  16
#define DIM_IN 1024
#define MROWS  2048   // both halves concatenated
#define NHALF  1024

// ---------------------------------------------------------------------------
// Feature guards (compile-safe: fall back when a builtin is absent; host pass
// sees __has_builtin(...) == 0 and parses only the plain-C fallbacks).
// ---------------------------------------------------------------------------
#if defined(__gfx1250__) && __has_builtin(__builtin_amdgcn_cvt_pk_bf16_f32)
#define HAVE_CVT_PK 1
#else
#define HAVE_CVT_PK 0
#endif

#if defined(__gfx1250__) && __has_builtin(__builtin_amdgcn_global_load_async_to_lds_b128)
#define HAVE_ASYNC_LDS 1
#else
#define HAVE_ASYNC_LDS 0
#endif

#if HAVE_ASYNC_LDS
// Builtin signature (from hipcc diagnostic): (v4i __device__* src,
// v4i __shared__* dst, imm int offset, imm int cpol)
typedef __attribute__((ext_vector_type(4))) int v4i;
typedef __attribute__((address_space(1))) v4i* gp_v4i;   // global int4*
typedef __attribute__((address_space(3))) v4i* lp_v4i;   // LDS int4*
#endif

// 16-byte global -> LDS copy; async (ASYNCcnt) when available.
__device__ __forceinline__ void copy16_g2l(const float* g, float* l) {
#if HAVE_ASYNC_LDS
  __builtin_amdgcn_global_load_async_to_lds_b128((gp_v4i)g, (lp_v4i)l, 0, 0);
#else
  *(float4*)l = *(const float4*)g;
#endif
}

// Wait for outstanding async LDS transfers (no-op on fallback path).
__device__ __forceinline__ void async_fence() {
#if HAVE_ASYNC_LDS
#if __has_builtin(__builtin_amdgcn_s_wait_asynccnt)
  __builtin_amdgcn_s_wait_asynccnt(0);
#else
  asm volatile("s_wait_asynccnt 0x0" ::: "memory");
#endif
#endif
}

// fp32 -> bf16: prefer the packed-convert builtin; otherwise use a plain
// fptrunc cast so the backend can select the native v_cvt_pk_bf16_f32-class
// instruction (instead of a hand-rolled RNE bit sequence).
#if HAVE_CVT_PK
typedef __attribute__((ext_vector_type(2))) __bf16 v2bf;
__device__ __forceinline__ v2bf cvt2_bf(float a, float b) {
  auto r = __builtin_amdgcn_cvt_pk_bf16_f32(a, b);
  return __builtin_bit_cast(v2bf, r);
}
__device__ __forceinline__ __bf16 f2bf(float f) { return cvt2_bf(f, f)[0]; }
#else
__device__ __forceinline__ __bf16 f2bf(float f) { return (__bf16)f; }
#endif

// Build a 16x32 bf16 A/B operand fragment from an fp32 row.
// CDNA5 16-bit operand layout: lane L (m=L%16, hi=L/16) holds
//   K = hi*8 + {0..7}      in elements 0..7
//   K = 16 + hi*8 + {0..7} in elements 8..15
__device__ __forceinline__ v16bf load_frag_f32(const float* __restrict__ row, int s0, int s1) {
  v16bf v;
#if HAVE_CVT_PK
#pragma unroll
  for (int i = 0; i < 4; ++i) {
    v2bf a = cvt2_bf(row[s0 + 2 * i], row[s0 + 2 * i + 1]);
    v2bf b = cvt2_bf(row[s1 + 2 * i], row[s1 + 2 * i + 1]);
    v[2 * i] = a[0]; v[2 * i + 1] = a[1];
    v[8 + 2 * i] = b[0]; v[9 + 2 * i] = b[1];
  }
#else
#pragma unroll
  for (int i = 0; i < 8; ++i) v[i] = (__bf16)row[s0 + i];
#pragma unroll
  for (int i = 0; i < 8; ++i) v[8 + i] = (__bf16)row[s1 + i];
#endif
  return v;
}

// Same fragment from bf16 memory (global or LDS); p0/p1 are the two
// 8-element contiguous segments (16B each).
__device__ __forceinline__ v16bf load_frag_bf(const __bf16* p0, const __bf16* p1) {
  v8bf a = *(const v8bf*)p0;
  v8bf b = *(const v8bf*)p1;
  v16bf v;
#pragma unroll
  for (int i = 0; i < 8; ++i) { v[i] = a[i]; v[8 + i] = b[i]; }
  return v;
}

// ---------------------------------------------------------------------------
// Kernel 1: three projection GEMMs sharing one shape:
//   q    = box_feats  @ Wq^T + Wq_b   -> bf16 [2048 x 1024], row-major
//   k    = part_feats @ Wk^T + Wk_b   -> bf16 [2048 x 1024], row-major
//   kp2T = (part_feats @ conv_w^T)^T  -> bf16 [1024 x 2048], TRANSPOSED
// Each wave owns one 16x16 output tile; K-loop of 32 per WMMA.
// ---------------------------------------------------------------------------
__global__ void __launch_bounds__(256)
proj_kernel(const float* __restrict__ bfeat, const float* __restrict__ pfeat,
            const float* __restrict__ Wq, const float* __restrict__ Wqb,
            const float* __restrict__ Wk, const float* __restrict__ Wkb,
            const float* __restrict__ Wc,
            unsigned short* __restrict__ q_raw,
            unsigned short* __restrict__ k_raw,
            unsigned short* __restrict__ kp2T_raw)
{
  __bf16* q    = (__bf16*)q_raw;
  __bf16* kk   = (__bf16*)k_raw;
  __bf16* kp2T = (__bf16*)kp2T_raw;

  const int lane  = threadIdx.x & 31;
  const int wave  = threadIdx.x >> 5;
  const int m     = lane & 15;
  const int hi    = lane >> 4;
  const int mtile = blockIdx.x * 8 + wave;   // 0..127
  const int jtile = blockIdx.y;              // 0..63
  const int mat   = blockIdx.z;              // 0:q 1:k 2:kp2T
  const int row0  = mtile * 16;
  const int col0  = jtile * 16;

  const float* A = (mat == 0) ? bfeat : pfeat;
  const float* W = (mat == 0) ? Wq : ((mat == 1) ? Wk : Wc);

  const float* arow = A + (size_t)(row0 + m) * DIM_IN;
  const float* wrow = W + (size_t)(col0 + m) * DIM_IN;

  v8f c = {};
  for (int kb = 0; kb < DIM_IN; kb += 32) {
    __builtin_prefetch(arow + kb + 128, 0, 1);
    __builtin_prefetch(wrow + kb + 128, 0, 1);
    const int s0 = kb + hi * 8;
    const int s1 = kb + 16 + hi * 8;
    v16bf a = load_frag_f32(arow, s0, s1);
    v16bf b = load_frag_f32(wrow, s0, s1);
    c = __builtin_amdgcn_wmma_f32_16x16x32_bf16(false, a, false, b, (short)0, c, false, false);
  }

  // C/D layout: VGPR r, lanes 0-15 -> M=r, N=lane ; lanes 16-31 -> M=r+8
  if (mat == 2) {
#pragma unroll
    for (int r = 0; r < 8; ++r) {
      const int row = row0 + r + hi * 8;
      const int col = col0 + m;
      kp2T[(size_t)col * MROWS + row] = f2bf(c[r]);   // transposed store (no bias)
    }
  } else {
    const float* bias = (mat == 0) ? Wqb : Wkb;
    const float bv = bias[col0 + m];
    __bf16* dst = (mat == 0) ? q : kk;
#pragma unroll
    for (int r = 0; r < 8; ++r) {
      const int row = row0 + r + hi * 8;
      const int col = col0 + m;
      dst[(size_t)row * DIM_IN + col] = f2bf(c[r] + bv);
    }
  }
}

// ---------------------------------------------------------------------------
// Kernel 2: fused position-embedding + attention + rel, flash-softmax style.
// Block = 16 n-rows of one half; 8 waves x 2 groups each; p-chunks of 32.
// ---------------------------------------------------------------------------
__global__ void __launch_bounds__(256)
attn_kernel(const float* __restrict__ rois, const float* __restrict__ prois,
            const float* __restrict__ Wg,   const float* __restrict__ Wgb,
            const float* __restrict__ convb,
            const unsigned short* __restrict__ q_raw,
            const unsigned short* __restrict__ k_raw,
            const unsigned short* __restrict__ kp2T_raw,
            float* __restrict__ out)
{
  const __bf16* q    = (const __bf16*)q_raw;
  const __bf16* kk   = (const __bf16*)k_raw;
  const __bf16* kp2T = (const __bf16*)kp2T_raw;

  __shared__ __align__(16) float sWg[GROUP * EM_DIM];
  __shared__ __align__(16) float sWgb[GROUP];
  __shared__ __align__(16) float sPr[32 * 5];      // part-ROI chunk (async-staged)
  __shared__ float sBox[16][4];                    // cx, cy, bw, bh
  __shared__ float sLogw[32 * 16 * GROUP];         // [p][n][g]
  __shared__ __align__(16) __bf16 sProbs[8][2][16 * 32];  // per (wave, group)

  const int tid  = threadIdx.x;
  const int lane = tid & 31;
  const int wave = tid >> 5;
  const int m    = lane & 15;
  const int hi   = lane >> 4;

  const int ntile = blockIdx.x;          // 0..63
  const int half  = blockIdx.y;          // 0..1  (n1_boxes == n1_parts == 1024)
  const int nrow0 = half * NHALF + ntile * 16;
  const int prow0 = half * NHALF;

  // Wg (4KB) + Wgb (64B) via async global->LDS DMA (ASYNCcnt path)
  if (tid < 256) copy16_g2l(Wg + tid * 4, sWg + tid * 4);
  if (tid < 4)   copy16_g2l(Wgb + tid * 4, sWgb + tid * 4);
  if (tid < 16) {
    const float* r = rois + (size_t)(nrow0 + tid) * 5;
    const float xmin = r[1], ymin = r[2], xmax = r[3], ymax = r[4];
    sBox[tid][0] = 0.5f * (xmin + xmax);
    sBox[tid][1] = 0.5f * (ymin + ymax);
    sBox[tid][2] = xmax - xmin + 1.0f;
    sBox[tid][3] = ymax - ymin + 1.0f;
  }
  async_fence();
  __syncthreads();

  const int g0 = wave * 2;

  // Preload q A-operand fragments: per group, K=64 split into 2 frags of 32.
  v16bf aq[2][2];
#pragma unroll
  for (int gi = 0; gi < 2; ++gi) {
    const __bf16* qrow = q + (size_t)(nrow0 + m) * DIM_IN + (g0 + gi) * EM_DIM;
#pragma unroll
    for (int f = 0; f < 2; ++f)
      aq[gi][f] = load_frag_bf(qrow + f * 32 + hi * 8, qrow + f * 32 + 16 + hi * 8);
  }

  v8f acc[2][4];
  float mrun[2][8], lrun[2][8];
  v8f zero = {};
#pragma unroll
  for (int gi = 0; gi < 2; ++gi) {
#pragma unroll
    for (int o = 0; o < 4; ++o) acc[gi][o] = zero;
#pragma unroll
    for (int r = 0; r < 8; ++r) { mrun[gi][r] = -1e30f; lrun[gi][r] = 0.0f; }
  }

  // 1 / 1000^(f/8)
  const float inv_em[8] = {1.0f, 0.42169650f, 0.17782794f, 0.074989421f,
                           0.031622777f, 0.013335214f, 0.0056234133f, 0.0023713737f};

  for (int pc = 0; pc < 32; ++pc) {
    const int pbase = pc * 32;

    __syncthreads();   // previous chunk fully consumed before overwriting LDS

    // stage this chunk's 32 part-ROIs (640B, 16B-aligned) into LDS
    if (tid < 40)
      copy16_g2l(prois + (size_t)(prow0 + pbase) * 5 + tid * 4, sPr + tid * 4);
    async_fence();
    __syncthreads();

    // ---- cooperative logw tile: 32p x 16n x 16g, 2 (n,p) pairs per thread
#pragma unroll
    for (int rep = 0; rep < 2; ++rep) {
      const int pair = tid + rep * 256;
      const int n = pair & 15;
      const int p = pair >> 4;             // 0..31
      const float pxmin = sPr[p * 5 + 1], pymin = sPr[p * 5 + 2];
      const float pxmax = sPr[p * 5 + 3], pymax = sPr[p * 5 + 4];
      const float pcx = 0.5f * (pxmin + pxmax), pcy = 0.5f * (pymin + pymax);
      const float pw = pxmax - pxmin + 1.0f, ph = pymax - pymin + 1.0f;
      const float cx = sBox[n][0], cy = sBox[n][1], bw = sBox[n][2], bh = sBox[n][3];
      float pos[4];
      pos[0] = __logf(fmaxf(fabsf((cx - pcx) / bw), 0.001f));
      pos[1] = __logf(fmaxf(fabsf((cy - pcy) / bh), 0.001f));
      pos[2] = __logf(pw / bw);
      pos[3] = __logf(ph / bh);
      float s[GROUP];
#pragma unroll
      for (int g = 0; g < GROUP; ++g) s[g] = sWgb[g];
#pragma unroll
      for (int j = 0; j < 4; ++j) {
#pragma unroll
        for (int f = 0; f < 8; ++f) {
          float sn, cs;
          __sincosf(100.0f * pos[j] * inv_em[f], &sn, &cs);
          const int e = j * 16 + f;        // emb index: sin at e, cos at e+8
#pragma unroll
          for (int g = 0; g < GROUP; ++g)
            s[g] += sn * sWg[g * EM_DIM + e] + cs * sWg[g * EM_DIM + e + 8];
        }
      }
#pragma unroll
      for (int g = 0; g < GROUP; ++g)
        sLogw[(p * 16 + n) * GROUP + g] = __logf(fmaxf(s[g], 1e-6f));
    }
    __syncthreads();

    // ---- per-wave flash step over this 32-wide p chunk
#pragma unroll
    for (int gi = 0; gi < 2; ++gi) {
      const int g = g0 + gi;
      v8f lt[2];
#pragma unroll
      for (int ps = 0; ps < 2; ++ps) {
        const int prow = prow0 + pbase + ps * 16 + m;        // B column = p
        const __bf16* krow = kk + (size_t)prow * DIM_IN + g * EM_DIM;
        v16bf b0 = load_frag_bf(krow + hi * 8,      krow + 16 + hi * 8);
        v16bf b1 = load_frag_bf(krow + 32 + hi * 8, krow + 48 + hi * 8);
        v8f t = {};
        t = __builtin_amdgcn_wmma_f32_16x16x32_bf16(false, aq[gi][0], false, b0, (short)0, t, false, false);
        t = __builtin_amdgcn_wmma_f32_16x16x32_bf16(false, aq[gi][1], false, b1, (short)0, t, false, false);
        const int pl = ps * 16 + m;                          // p within chunk (N=lane%16)
#pragma unroll
        for (int r = 0; r < 8; ++r) {
          const int nl = r + hi * 8;                         // n within tile
          lt[ps][r] = t[r] * (1.0f / GROUP) + sLogw[(pl * 16 + nl) * GROUP + g];
        }
      }
      // online softmax: row max / rescale / row sum (reductions stay inside
      // each 16-lane half, matching the C-tile row mapping)
#pragma unroll
      for (int r = 0; r < 8; ++r) {
        float v = fmaxf(lt[0][r], lt[1][r]);
        v = fmaxf(v, __shfl_xor(v, 1));
        v = fmaxf(v, __shfl_xor(v, 2));
        v = fmaxf(v, __shfl_xor(v, 4));
        v = fmaxf(v, __shfl_xor(v, 8));
        const float nm = fmaxf(mrun[gi][r], v);
        const float alpha = __expf(mrun[gi][r] - nm);
        mrun[gi][r] = nm;
        const float e0 = __expf(lt[0][r] - nm);
        const float e1 = __expf(lt[1][r] - nm);
        lt[0][r] = e0; lt[1][r] = e1;
        float rs = e0 + e1;
        rs += __shfl_xor(rs, 1);
        rs += __shfl_xor(rs, 2);
        rs += __shfl_xor(rs, 4);
        rs += __shfl_xor(rs, 8);
        lrun[gi][r] = lrun[gi][r] * alpha + rs;
#pragma unroll
        for (int o = 0; o < 4; ++o) acc[gi][o][r] *= alpha;
      }
      // stage probs through per-wave LDS to re-layout C-tile -> A operand
      __bf16* pb = &sProbs[wave][gi][0];
#pragma unroll
      for (int ps = 0; ps < 2; ++ps) {
        const int pl = ps * 16 + m;
#pragma unroll
        for (int r = 0; r < 8; ++r)
          pb[(r + hi * 8) * 32 + pl] = f2bf(lt[ps][r]);
      }
      // same-wave LDS ops are in-order (DScnt); compiler inserts the wait
      v16bf pa = load_frag_bf(pb + m * 32 + hi * 8, pb + m * 32 + 16 + hi * 8);

      // rel += probs(16x32) @ kp2(32x64): 4 output tiles of 16 cols
#pragma unroll
      for (int o = 0; o < 4; ++o) {
        const __bf16* kcol = kp2T + (size_t)(g * EM_DIM + o * 16 + m) * MROWS
                                  + prow0 + pbase;           // contiguous in p
        v16bf bo = load_frag_bf(kcol + hi * 8, kcol + 16 + hi * 8);
        acc[gi][o] = __builtin_amdgcn_wmma_f32_16x16x32_bf16(false, pa, false, bo, (short)0, acc[gi][o], false, false);
      }
    }
  }

  // epilogue: normalize, add conv bias (softmax rows sum to 1, so conv_b
  // passes through the fused (sm @ (pf@w3^T)) re-association unchanged)
#pragma unroll
  for (int gi = 0; gi < 2; ++gi) {
#pragma unroll
    for (int o = 0; o < 4; ++o) {
      const int col = (g0 + gi) * EM_DIM + o * 16 + m;
      const float cb = convb[col];
#pragma unroll
      for (int r = 0; r < 8; ++r) {
        const int row = nrow0 + r + hi * 8;
        out[(size_t)row * DIM_IN + col] = acc[gi][o][r] / lrun[gi][r] + cb;
      }
    }
  }
}

extern "C" void kernel_launch(void* const* d_in, const int* in_sizes, int n_in,
                              void* d_out, int out_size, void* d_ws, size_t ws_size,
                              hipStream_t stream) {
  (void)in_sizes; (void)n_in; (void)out_size; (void)ws_size;
  const float* rois  = (const float*)d_in[0];
  const float* prois = (const float*)d_in[1];
  const float* bfeat = (const float*)d_in[2];
  const float* pfeat = (const float*)d_in[3];
  const float* Wg    = (const float*)d_in[4];
  const float* Wgb   = (const float*)d_in[5];
  const float* Wq    = (const float*)d_in[6];
  const float* Wqb   = (const float*)d_in[7];
  const float* Wk    = (const float*)d_in[8];
  const float* Wkb   = (const float*)d_in[9];
  const float* Wc    = (const float*)d_in[10];
  const float* convb = (const float*)d_in[11];
  // d_in[12], d_in[13]: n1_boxes = n1_parts = 1024 (fixed by setup_inputs)

  // workspace: q(4MB bf16) | k(4MB bf16) | kp2T(4MB bf16, transposed)
  unsigned short* qws  = (unsigned short*)d_ws;
  unsigned short* kws  = qws + (size_t)MROWS * DIM_IN;
  unsigned short* kp2T = kws + (size_t)MROWS * DIM_IN;

  proj_kernel<<<dim3(16, 64, 3), 256, 0, stream>>>(
      bfeat, pfeat, Wq, Wqb, Wk, Wkb, Wc, qws, kws, kp2T);

  attn_kernel<<<dim3(64, 2), 256, 0, stream>>>(
      rois, prois, Wg, Wgb, convb, qws, kws, kp2T, (float*)d_out);
}
